// E2FastPredictor_9826885173440
// MI455X (gfx1250) — matrix operand order repaired and checked by reading, not compiled, not run on hardware
//
#include <hip/hip_runtime.h>

typedef __bf16 bf16;
typedef __attribute__((ext_vector_type(16))) __bf16 v16bf;
typedef __attribute__((ext_vector_type(8)))  __bf16 v8bf;
typedef __attribute__((ext_vector_type(8)))  float  v8f;
typedef __attribute__((ext_vector_type(4)))  unsigned int v4u;
typedef __attribute__((ext_vector_type(4)))  int v4i;

#define AS_GLOBAL __attribute__((address_space(1)))
#define AS_LOCAL  __attribute__((address_space(3)))

#if defined(__HIP_DEVICE_COMPILE__) && __has_builtin(__builtin_amdgcn_global_load_async_to_lds_b128)
#define HAVE_ASYNC_LDS 1
#else
#define HAVE_ASYNC_LDS 0
#endif

static constexpr int Bsz = 4096;
static constexpr int HOR = 32;
static constexpr int Ldim = 512;
static constexpr int Adim = 64;
static constexpr int HID = 1024;

// LDS stage geometry: 128 rows x 64 bf16 per stage, padded to 72 elems/row
static constexpr int KC   = 64;   // K elements per stage (2 WMMA K-steps)
static constexpr int LPAD = 72;   // row stride in bf16 elements (144B: conflict-free)

// Epilogue modes
enum { M_BF16 = 0, M_BF16_RELU = 1, M_F32_RELU = 2, M_F32 = 3, M_NEXT = 4 };

__device__ __forceinline__ bf16 tobf(float f) { return (bf16)f; }

__device__ __forceinline__ void wait_async0() {
#if HAVE_ASYNC_LDS
#if __has_builtin(__builtin_amdgcn_s_wait_asynccnt)
    __builtin_amdgcn_s_wait_asynccnt(0);
#else
    asm volatile("s_wait_asynccnt 0x0" ::: "memory");
#endif
#endif
}

// Issue one A-tile stage (128 rows x 64 bf16 = 16KB) into an LDS buffer.
// 256 threads x 4 chunks of 16B each. Async (ASYNCcnt) when available.
__device__ __forceinline__ void stage_issue(const bf16* __restrict__ A, int lda, int row0,
                                            int kt, bf16 (*buf)[LPAD], int tid) {
#pragma unroll
    for (int i = 0; i < 4; ++i) {
        int idx = tid + i * 256;
        int rr  = idx >> 3;          // 0..127
        int sg  = idx & 7;           // 8 x 16B segments per row
        const bf16* g = A + (size_t)(row0 + rr) * lda + kt * KC + sg * 8;
        bf16* l = &buf[rr][sg * 8];
#if HAVE_ASYNC_LDS
        __builtin_amdgcn_global_load_async_to_lds_b128(
            (AS_GLOBAL v4i*)g, (AS_LOCAL v4i*)l, 0, 0);
#else
        *(v4u*)l = *(const v4u*)g;
#endif
    }
}

// ---------------------------------------------------------------------------
// Pack a row-major f32 weight [K, N] into bf16 WMMA B-fragments:
//   dst[(kt*(N/16)+nt)*512 + lane*16 + i] = W[(kt*32 + (lane>=16?8:0) + i + (i>=8?8:0))*N + nt*16 + (lane&15)]
// One wave per 32x16 fragment.
// ---------------------------------------------------------------------------
__global__ void pack_b_kernel(const float* __restrict__ W, bf16* __restrict__ dst,
                              int K, int N) {
    int fr   = blockIdx.x * (blockDim.x >> 5) + (threadIdx.x >> 5);
    int lane = threadIdx.x & 31;
    int nfr  = (K >> 5) * (N >> 4);
    if (fr >= nfr) return;
    int ntiles = N >> 4;
    int kt = fr / ntiles;
    int nt = fr - kt * ntiles;
    int kb = kt * 32 + ((lane >> 4) << 3);
    int nc = nt * 16 + (lane & 15);
    bf16* d = dst + (size_t)fr * 512 + lane * 16;
#pragma unroll
    for (int i = 0; i < 16; ++i)
        d[i] = tobf(W[(size_t)(kb + i + (i >= 8 ? 8 : 0)) * N + nc]);
}

// f32 -> bf16 elementwise
__global__ void cvt_bf16_kernel(const float* __restrict__ src, bf16* __restrict__ dst, int n) {
    int i = blockIdx.x * blockDim.x + threadIdx.x;
    if (i < n) dst[i] = tobf(src[i]);
}

// initial state -> stateF, stateB, out_states[:,0,:]
__global__ void init_state_kernel(const float* __restrict__ src, float* __restrict__ sF,
                                  bf16* __restrict__ sB, float* __restrict__ outStates) {
    int i = blockIdx.x * blockDim.x + threadIdx.x;   // i < B*L
    float v = src[i];
    sF[i] = v;
    sB[i] = tobf(v);
    int b = i >> 9;           // /512
    int n = i & 511;
    outStates[(size_t)b * ((HOR + 1) * Ldim) + n] = v;
}

// harm head: sigmoid(dot(hh[b,:1024], Wh2) + bh2) -> out[b*HOR + t]
__global__ void harm_kernel(const float* __restrict__ hh, const float* __restrict__ w,
                            const float* __restrict__ b2, float* __restrict__ outH, int t) {
    int row  = blockIdx.x * (blockDim.x >> 5) + (threadIdx.x >> 5);
    int lane = threadIdx.x & 31;
    const float* hr = hh + (size_t)row * HID;
    float s = 0.f;
#pragma unroll 8
    for (int i = 0; i < 32; ++i)
        s += hr[i * 32 + lane] * w[i * 32 + lane];
#pragma unroll
    for (int off = 16; off > 0; off >>= 1)
        s += __shfl_down(s, off, 32);
    if (lane == 0) {
        float x = s + b2[0];
        outH[(size_t)row * HOR + t] = 1.0f / (1.0f + expf(-x));
    }
}

// ---------------------------------------------------------------------------
// Tiled WMMA GEMM:  C[M,N] = A1[M,K1]@Bp1 (+ A2[M,K2]@Bp2) + bias, fused epilogue.
//   A row-major bf16 (lda in elements); Bp packed fragments (see pack_b_kernel).
//   Block = 256 threads (8 waves), tile 128(M) x 64(N); wave = 2x2 16x16 WMMAs.
//   K multiple of 64. If A2 != null, K1/64 must be even (buffer-parity safety);
//   all call sites satisfy this (K1 = 512).
//   Double-buffered LDS staging; async memory->LDS path when toolchain has it.
// ---------------------------------------------------------------------------
template <int MODE>
__global__ __launch_bounds__(256) void gemm_wmma(
    const bf16* __restrict__ A1, int lda1, int K1, const bf16* __restrict__ Bp1,
    const bf16* __restrict__ A2, int lda2, int K2, const bf16* __restrict__ Bp2,
    const float* __restrict__ bias, int N,
    float* __restrict__ outF, int ldoF,
    bf16* __restrict__ outB, int ldoB,
    float* __restrict__ stateF, bf16* __restrict__ stateB) {

    __shared__ __align__(16) bf16 As[2][128][LPAD];   // 36 KB, double buffered

    const int tid  = threadIdx.x;
    const int lane = tid & 31;
    const int wave = tid >> 5;
    const int wm   = wave & 3;                   // 0..3  (M direction, 32 rows each)
    const int wn   = wave >> 2;                  // 0..1  (N direction, 32 cols each)
    const int row0 = blockIdx.y * 128;
    const int ntile0 = (blockIdx.x * 64 + wn * 32) >> 4;  // first 16-wide N tile index
    const int mhalf  = lane & 15;
    const int kbase  = (lane >> 4) << 3;         // 0 or 8 (A-fragment K base, elements)

    v8f acc00 = {}, acc01 = {}, acc10 = {}, acc11 = {};

    for (int s = 0; s < 2; ++s) {
        const bf16* A  = s ? A2 : A1;
        const bf16* Bp = s ? Bp2 : Bp1;
        if (A == nullptr) continue;              // uniform across grid
        const int lda = s ? lda2 : lda1;
        const int K   = s ? K2 : K1;
        const int nst = K >> 6;                  // 64-wide stages

        stage_issue(A, lda, row0, 0, As[0], tid);   // prologue

        for (int kt = 0; kt < nst; ++kt) {
            wait_async0();        // stage kt resident in LDS (this wave's issues)
            __syncthreads();      // all waves: stage kt visible; prev buffer free
            if (kt + 1 < nst)     // overlap: stream stage kt+1 into other buffer
                stage_issue(A, lda, row0, kt + 1, As[(kt + 1) & 1], tid);

            const bf16(*cbuf)[LPAD] = As[kt & 1];

            if (kt + 1 < nst) {   // hint next stage's B fragments toward L2/L0
                const bf16* pf = Bp + ((size_t)(kt + 1) * 2 * (N >> 4) + ntile0) * 512 + lane * 16;
                __builtin_prefetch(pf, 0, 0);
            }

#pragma unroll
            for (int kc = 0; kc < 2; ++kc) {     // two 32-deep WMMA K-steps per stage
                // A fragments per ISA 16-bit A layout: lane m=lane&15,
                // elems i<8 -> K=kbase+i, i>=8 -> K=16+kbase+(i-8)
                const bf16* r0 = &cbuf[wm * 32 + mhalf][kc * 32];
                const bf16* r1 = &cbuf[wm * 32 + 16 + mhalf][kc * 32];
                v8bf a0lo = *(const v8bf*)(r0 + kbase);
                v8bf a0hi = *(const v8bf*)(r0 + 16 + kbase);
                v8bf a1lo = *(const v8bf*)(r1 + kbase);
                v8bf a1hi = *(const v8bf*)(r1 + 16 + kbase);
                v16bf a0 = __builtin_shufflevector(a0lo, a0hi, 0,1,2,3,4,5,6,7,8,9,10,11,12,13,14,15);
                v16bf a1 = __builtin_shufflevector(a1lo, a1hi, 0,1,2,3,4,5,6,7,8,9,10,11,12,13,14,15);

                // B fragments: contiguous 32B per lane from packed (L2-resident) weights
                const int kt2 = kt * 2 + kc;     // 32-deep tile index
                const bf16* bb = Bp + ((size_t)kt2 * (N >> 4) + ntile0) * 512 + lane * 16;
                v16bf b0 = *(const v16bf*)(bb);
                v16bf b1 = *(const v16bf*)(bb + 512);

                acc00 = __builtin_amdgcn_wmma_f32_16x16x32_bf16(false, a0, false, b0, (short)0, acc00, false, false);
                acc01 = __builtin_amdgcn_wmma_f32_16x16x32_bf16(false, a0, false, b1, (short)0, acc01, false, false);
                acc10 = __builtin_amdgcn_wmma_f32_16x16x32_bf16(false, a1, false, b0, (short)0, acc10, false, false);
                acc11 = __builtin_amdgcn_wmma_f32_16x16x32_bf16(false, a1, false, b1, (short)0, acc11, false, false);
            }
        }
    }

    // Epilogue. C/D layout: VGPR r, lane l -> M = r + (l>=16 ? 8:0), N = l&15.
    const int rbase = (lane >> 4) << 3;
#pragma unroll
    for (int si = 0; si < 2; ++si) {
#pragma unroll
        for (int sj = 0; sj < 2; ++sj) {
            v8f acc = (si == 0) ? ((sj == 0) ? acc00 : acc01)
                                : ((sj == 0) ? acc10 : acc11);
            int col = blockIdx.x * 64 + wn * 32 + sj * 16 + (lane & 15);
            float bv = bias[col];
            int rowt = row0 + wm * 32 + si * 16 + rbase;
#pragma unroll
            for (int r = 0; r < 8; ++r) {
                int row = rowt + r;
                float v = acc[r] + bv;
                if (MODE == M_BF16_RELU || MODE == M_F32_RELU) v = fmaxf(v, 0.0f);
                if (MODE == M_BF16 || MODE == M_BF16_RELU)
                    outB[(size_t)row * ldoB + col] = tobf(v);
                if (MODE == M_F32_RELU || MODE == M_F32)
                    outF[(size_t)row * ldoF + col] = v;
                if (MODE == M_NEXT) {
                    size_t sidx = (size_t)row * Ldim + col;
                    v += stateF[sidx];
                    outF[(size_t)row * ldoF + col] = v;
                    stateF[sidx] = v;
                    stateB[sidx] = tobf(v);
                }
            }
        }
    }
}

// ---------------------------------------------------------------------------
// Host-side orchestration
// ---------------------------------------------------------------------------
extern "C" void kernel_launch(void* const* d_in, const int* in_sizes, int n_in,
                              void* d_out, int out_size, void* d_ws, size_t ws_size,
                              hipStream_t stream) {
    const float* initial_state = (const float*)d_in[0];
    const float* action_seq    = (const float*)d_in[1];
    const float* Wae = (const float*)d_in[2];  const float* bae = (const float*)d_in[3];
    const float* Wt1 = (const float*)d_in[4];  const float* bt1 = (const float*)d_in[5];
    const float* Wt2 = (const float*)d_in[6];  const float* bt2 = (const float*)d_in[7];
    const float* Wt3 = (const float*)d_in[8];  const float* bt3 = (const float*)d_in[9];
    const float* Wo1 = (const float*)d_in[10]; const float* bo1 = (const float*)d_in[11];
    const float* Wo2 = (const float*)d_in[12]; const float* bo2 = (const float*)d_in[13];
    const float* Wh1 = (const float*)d_in[14]; const float* bh1 = (const float*)d_in[15];
    const float* Wh2 = (const float*)d_in[16]; const float* bh2 = (const float*)d_in[17];

    float* out_states = (float*)d_out;                                   // [B, 33, 512]
    float* out_obs    = out_states + (size_t)Bsz * (HOR + 1) * Ldim;     // [B, 32, 512]
    float* out_harm   = out_obs + (size_t)Bsz * HOR * Ldim;              // [B, 32]

    char*  ws  = (char*)d_ws;
    size_t off = 0;
    auto alloc = [&](size_t bytes) -> char* {
        char* p = ws + off;
        off += (bytes + 255) & ~(size_t)255;
        return p;
    };
    bf16* pWh1s = (bf16*)alloc((size_t)512 * 1024 * 2);
    bf16* pWh1a = (bf16*)alloc((size_t)64 * 1024 * 2);
    bf16* pWo1  = (bf16*)alloc((size_t)512 * 1024 * 2);
    bf16* pWo2  = (bf16*)alloc((size_t)1024 * 512 * 2);
    bf16* pWae  = (bf16*)alloc((size_t)64 * 64 * 2);
    bf16* pWt1s = (bf16*)alloc((size_t)512 * 1024 * 2);
    bf16* pWt1a = (bf16*)alloc((size_t)64 * 1024 * 2);
    bf16* pWt2  = (bf16*)alloc((size_t)1024 * 1024 * 2);
    bf16* pWt3  = (bf16*)alloc((size_t)1024 * 512 * 2);
    float* stateF = (float*)alloc((size_t)Bsz * Ldim * 4);
    bf16*  stateB = (bf16*)alloc((size_t)Bsz * Ldim * 2);
    bf16*  actB   = (bf16*)alloc((size_t)Bsz * HOR * Adim * 2);
    bf16*  aeB    = (bf16*)alloc((size_t)Bsz * Adim * 2);
    bf16*  h1B    = (bf16*)alloc((size_t)Bsz * HID * 2);
    bf16*  h2B    = (bf16*)alloc((size_t)Bsz * HID * 2);
    float* hhF    = (float*)alloc((size_t)Bsz * HID * 4);

    auto pack = [&](const float* W, bf16* dst, int K, int N) {
        int frags  = (K >> 5) * (N >> 4);
        int blocks = (frags + 7) / 8;
        pack_b_kernel<<<blocks, 256, 0, stream>>>(W, dst, K, N);
    };
    pack(Wh1,                pWh1s, 512,  1024);
    pack(Wh1 + 512 * 1024,   pWh1a, 64,   1024);
    pack(Wo1,                pWo1,  512,  1024);
    pack(Wo2,                pWo2,  1024, 512);
    pack(Wae,                pWae,  64,   64);
    pack(Wt1,                pWt1s, 512,  1024);
    pack(Wt1 + 512 * 1024,   pWt1a, 64,   1024);
    pack(Wt2,                pWt2,  1024, 1024);
    pack(Wt3,                pWt3,  1024, 512);

    {   // actions -> bf16, state init
        int n = Bsz * HOR * Adim;
        cvt_bf16_kernel<<<(n + 255) / 256, 256, 0, stream>>>(action_seq, actB, n);
        int m = Bsz * Ldim;
        init_state_kernel<<<(m + 255) / 256, 256, 0, stream>>>(initial_state, stateF, stateB, out_states);
    }

    const dim3 blk(256);
    const dim3 gN1024(1024 / 64, Bsz / 128);
    const dim3 gN512(512 / 64, Bsz / 128);
    const dim3 gN64(64 / 64, Bsz / 128);
    const int ldAct = HOR * Adim;   // action row stride in the [B,HOR,A] tensor

    for (int t = 0; t < HOR; ++t) {
        const bf16* actT = actB + (size_t)t * Adim;

        // harm hidden: relu(state@Wh1s + action@Wh1a + bh1) -> hhF [B,1024] f32
        gemm_wmma<M_F32_RELU><<<gN1024, blk, 0, stream>>>(
            stateB, Ldim, Ldim, pWh1s, actT, ldAct, Adim, pWh1a,
            bh1, HID, hhF, HID, nullptr, 0, nullptr, nullptr);
        // harm: sigmoid(hh @ Wh2 + bh2) -> out_harm[:, t]
        harm_kernel<<<Bsz / 8, blk, 0, stream>>>(hhF, Wh2, bh2, out_harm, t);

        // obs: relu(state@Wo1 + bo1) -> h1B; h1B@Wo2 + bo2 -> out_obs[:, t, :]
        gemm_wmma<M_BF16_RELU><<<gN1024, blk, 0, stream>>>(
            stateB, Ldim, Ldim, pWo1, nullptr, 0, 0, nullptr,
            bo1, HID, nullptr, 0, h1B, HID, nullptr, nullptr);
        gemm_wmma<M_F32><<<gN512, blk, 0, stream>>>(
            h1B, HID, HID, pWo2, nullptr, 0, 0, nullptr,
            bo2, Ldim, out_obs + (size_t)t * Ldim, HOR * Ldim, nullptr, 0, nullptr, nullptr);

        // ae = action@Wae + bae -> aeB [B,64]
        gemm_wmma<M_BF16><<<gN64, blk, 0, stream>>>(
            actT, ldAct, Adim, pWae, nullptr, 0, 0, nullptr,
            bae, Adim, nullptr, 0, aeB, Adim, nullptr, nullptr);

        // t1 = relu(state@Wt1s + ae@Wt1a + bt1) -> h1B
        gemm_wmma<M_BF16_RELU><<<gN1024, blk, 0, stream>>>(
            stateB, Ldim, Ldim, pWt1s, aeB, Adim, Adim, pWt1a,
            bt1, HID, nullptr, 0, h1B, HID, nullptr, nullptr);
        // t2 = relu(t1@Wt2 + bt2) -> h2B
        gemm_wmma<M_BF16_RELU><<<gN1024, blk, 0, stream>>>(
            h1B, HID, HID, pWt2, nullptr, 0, 0, nullptr,
            bt2, HID, nullptr, 0, h2B, HID, nullptr, nullptr);
        // next = state + t2@Wt3 + bt3 -> out_states[:, t+1, :], stateF, stateB
        gemm_wmma<M_NEXT><<<gN512, blk, 0, stream>>>(
            h2B, HID, HID, pWt3, nullptr, 0, 0, nullptr,
            bt3, Ldim, out_states + (size_t)(t + 1) * Ldim, (HOR + 1) * Ldim,
            nullptr, 0, stateF, stateB);
    }
}
